// CoNHDNodeScorer_87282325389911
// MI455X (gfx1250) — matrix operand order repaired and checked by reading, not compiled, not run on hardware
//
#include <hip/hip_runtime.h>
#include <hip/hip_bf16.h>

typedef _Float16 v16h __attribute__((ext_vector_type(16)));
typedef _Float16 v8h  __attribute__((ext_vector_type(8)));
typedef float    v8f  __attribute__((ext_vector_type(8)));

static constexpr int N_NODES = 50000;
static constexpr int N_EDGES = 800000;
static constexpr int DIM     = 128;   // CO_REP_DIM
static constexpr int HID     = 256;   // HIDDEN
static constexpr int NCLS    = 40;    // NUM_CLASSES

union AV { v16h v; v8h h[2]; };

// ---------------------------------------------------------------- zero ws
__global__ void zero_ws_kernel(unsigned* __restrict__ p, int n) {
  int i = blockIdx.x * 256 + threadIdx.x;
  if (i < n) p[i] = 0u;
}

// ------------------------------------------------- gather + scatter-add
// One wave (32 lanes) per edge; each lane moves 4 contiguous floats of the
// 128-float row and accumulates into sums[dst] with hardware f32 atomics
// (explicit CDNA5 global_atomic_add_f32, non-returning -> STOREcnt; the
// implicit wait-idle at s_endpgm covers completion). Counts via u32 atomic.
__global__ void scatter_kernel(const float* __restrict__ co_feat,
                               const int*   __restrict__ co_idx,
                               const int*   __restrict__ edge_dst,
                               float*       __restrict__ sums,
                               unsigned*    __restrict__ counts) {
  unsigned gid  = blockIdx.x * 256u + threadIdx.x;
  unsigned e    = gid >> 5;
  unsigned lane = gid & 31u;
  if (e >= (unsigned)N_EDGES) return;
  int idx = co_idx[e];
  int dst = edge_dst[e];
  const float4* src = (const float4*)(co_feat + (size_t)idx * DIM);
  float4 g = src[lane];
  float* s = sums + (size_t)dst * DIM + lane * 4;
  asm volatile("global_atomic_add_f32 %0, %1, off"          :: "v"(s), "v"(g.x) : "memory");
  asm volatile("global_atomic_add_f32 %0, %1, off offset:4" :: "v"(s), "v"(g.y) : "memory");
  asm volatile("global_atomic_add_f32 %0, %1, off offset:8" :: "v"(s), "v"(g.z) : "memory");
  asm volatile("global_atomic_add_f32 %0, %1, off offset:12":: "v"(s), "v"(g.w) : "memory");
  if (lane == 0) atomicAdd(counts + dst, 1u);
}

// ------------------------------------------- mean + f32 -> f16 convert
__global__ void normalize_kernel(const float* __restrict__ sums,
                                 const unsigned* __restrict__ counts,
                                 _Float16* __restrict__ vh) {
  int i = blockIdx.x * 256 + threadIdx.x;
  if (i >= N_NODES * DIM) return;
  unsigned c = counts[i >> 7];
  float d = (c > 0u) ? (float)c : 1.0f;
  vh[i] = (_Float16)(sums[i] / d);
}

// --------------------- repack W1 [128,256] into WMMA B-operand layout
// Tile (kc,nc): 32x16 f16 block. Lane l holds col n = nc*16 + (l&15),
// element j holds K = kc*32 + (j&7) + 8*(l>=16) + 16*(j>=8). Stored contiguous per lane.
__global__ void prep_w1_kernel(const float* __restrict__ W1,
                               _Float16* __restrict__ W1B) {
  int gid = blockIdx.x * 256 + threadIdx.x;
  if (gid >= 4 * 16 * 32 * 16) return;
  int j  = gid & 15;
  int l  = (gid >> 4) & 31;
  int nc = (gid >> 9) & 15;
  int kc = gid >> 13;
  int k  = kc * 32 + (j & 7) + ((l >> 4) << 3) + ((j >> 3) << 4);
  int n  = nc * 16 + (l & 15);
  W1B[gid] = (_Float16)W1[k * HID + n];
}

// --------------------- repack W2 [256,40] (pad N->48) into B layout
__global__ void prep_w2_kernel(const float* __restrict__ W2,
                               _Float16* __restrict__ W2B) {
  int gid = blockIdx.x * 256 + threadIdx.x;
  if (gid >= 8 * 3 * 32 * 16) return;
  int j  = gid & 15;
  int l  = (gid >> 4) & 31;
  int t  = gid >> 9;          // 0..23 = kc*3 + nc
  int nc = t % 3;
  int kc = t / 3;
  int k  = kc * 32 + (j & 7) + ((l >> 4) << 3) + ((j >> 3) << 4);
  int n  = nc * 16 + (l & 15);
  W2B[gid] = (n < NCLS) ? (_Float16)W2[k * NCLS + n] : (_Float16)0.0f;
}

// ------------------------------------------------------ fused WMMA MLP
// One wave per 16-node tile. GEMM1: h = relu(vfeat @ W1 + b1) staged in LDS
// as f16; GEMM2: out = h @ W2 + b2.
__global__ __launch_bounds__(32) void mlp_kernel(
    const _Float16* __restrict__ vh,
    const _Float16* __restrict__ W1B, const float* __restrict__ b1,
    const _Float16* __restrict__ W2B, const float* __restrict__ b2,
    float* __restrict__ out) {
  __shared__ __align__(32) _Float16 hbuf[16 * HID];   // 8 KB
  const int tile = blockIdx.x;
  const int l    = threadIdx.x;
  const int r    = l & 15;        // row (A) or col (B/C/D) within tile
  const int hi   = l >> 4;        // lane-half selector

  // A operand: vfeat rows, 16-bit A layout = two 8-half runs per K=32 chunk
  const _Float16* arow = vh + (size_t)(tile * 16 + r) * DIM;
  AV a[4];
#pragma unroll
  for (int kc = 0; kc < 4; ++kc) {
    a[kc].h[0] = *(const v8h*)(arow + kc * 32 + hi * 8);
    a[kc].h[1] = *(const v8h*)(arow + kc * 32 + hi * 8 + 16);
  }

  // GEMM1: 16 column tiles of HID=256, K=128 (4 WMMA each)
#pragma unroll
  for (int nc = 0; nc < 16; ++nc) {
    v8f acc = {};
#pragma unroll
    for (int kc = 0; kc < 4; ++kc) {
      v16h b = *(const v16h*)(W1B + (size_t)((kc * 16 + nc) * 32 + l) * 16);
      acc = __builtin_amdgcn_wmma_f32_16x16x32_f16(
          false, a[kc].v, false, b, (short)0, acc, false, false);
    }
    float bb = b1[nc * 16 + r];
#pragma unroll
    for (int v = 0; v < 8; ++v) {
      float x = acc[v] + bb;
      x = x > 0.0f ? x : 0.0f;
      hbuf[(v + hi * 8) * HID + nc * 16 + r] = (_Float16)x;  // row-major h
    }
  }
  __syncthreads();

  // GEMM2: 3 column tiles (40 classes padded to 48), K=256 (8 WMMA each)
#pragma unroll
  for (int nc2 = 0; nc2 < 3; ++nc2) {
    v8f acc = {};
#pragma unroll
    for (int kc2 = 0; kc2 < 8; ++kc2) {
      AV a2;
      a2.h[0] = *(const v8h*)&hbuf[r * HID + kc2 * 32 + hi * 8];
      a2.h[1] = *(const v8h*)&hbuf[r * HID + kc2 * 32 + hi * 8 + 16];
      v16h b = *(const v16h*)(W2B + (size_t)((kc2 * 3 + nc2) * 32 + l) * 16);
      acc = __builtin_amdgcn_wmma_f32_16x16x32_f16(
          false, a2.v, false, b, (short)0, acc, false, false);
    }
    int n = nc2 * 16 + r;
    if (n < NCLS) {
      float bb = b2[n];
#pragma unroll
      for (int v = 0; v < 8; ++v)
        out[(size_t)(tile * 16 + v + hi * 8) * NCLS + n] = acc[v] + bb;
    }
  }
}

// ---------------------------------------------------------------- launch
extern "C" void kernel_launch(void* const* d_in, const int* in_sizes, int n_in,
                              void* d_out, int out_size, void* d_ws, size_t ws_size,
                              hipStream_t stream) {
  const float* co_feat  = (const float*)d_in[0];
  const int*   co_idx   = (const int*)d_in[1];
  const int*   edge_dst = (const int*)d_in[2];
  const float* W1       = (const float*)d_in[3];
  const float* b1       = (const float*)d_in[4];
  const float* W2       = (const float*)d_in[5];
  const float* b2       = (const float*)d_in[6];
  float* out = (float*)d_out;

  // Workspace layout (all offsets 16/32-byte aligned):
  char* ws = (char*)d_ws;
  float*     sums   = (float*)ws;                          // 25,600,000 B
  unsigned*  counts = (unsigned*)(ws + 25600000);          //    200,000 B
  _Float16*  vh     = (_Float16*)(ws + 25800000);          // 12,800,000 B
  _Float16*  W1B    = (_Float16*)(ws + 38600000);          //     65,536 B
  _Float16*  W2B    = (_Float16*)(ws + 38665536);          //     24,576 B

  const int zero_n = N_NODES * DIM + N_NODES;              // sums + counts words
  zero_ws_kernel<<<(zero_n + 255) / 256, 256, 0, stream>>>((unsigned*)ws, zero_n);

  scatter_kernel<<<(N_EDGES * 32) / 256, 256, 0, stream>>>(
      co_feat, co_idx, edge_dst, sums, counts);

  normalize_kernel<<<(N_NODES * DIM) / 256, 256, 0, stream>>>(sums, counts, vh);

  prep_w1_kernel<<<(4 * 16 * 32 * 16) / 256, 256, 0, stream>>>(W1, W1B);
  prep_w2_kernel<<<(8 * 3 * 32 * 16) / 256, 256, 0, stream>>>(W2, W2B);

  mlp_kernel<<<N_NODES / 16, 32, 0, stream>>>(vh, W1B, b1, W2B, b2, out);
}
